// GraphDownsample_47038481825902
// MI455X (gfx1250) — matrix operand order repaired and checked by reading, not compiled, not run on hardware
//
#include <hip/hip_runtime.h>
#include <hip/hip_bf16.h>

typedef __attribute__((ext_vector_type(16))) __bf16       v16bf;
typedef __attribute__((ext_vector_type(8)))  float        v8f;
typedef __attribute__((ext_vector_type(8)))  unsigned int u32x8;
typedef __attribute__((ext_vector_type(4)))  int          i32x4;

#define PREFIX_N   20000
#define LEAF_N     50000
#define NONLEAF_N  50000
#define CDIM       256
#define KDIM       2048      // 8*C
#define MGEMM      50000     // NUMD/8
#define NGEMM      256
#define KSTEPS     64        // KDIM/32
#define MBLK       64
#define APITCH     40        // ushorts per LDS row (32 data + 8 pad) -> 80B, conflict-free
#define BPITCH     40
#define CPITCH     260       // floats per epilogue LDS row (256 + 4 pad), 16B-aligned pitch

__device__ __forceinline__ unsigned short bf16_trunc(float f) {
  unsigned u = __builtin_bit_cast(unsigned, f);
  return (unsigned short)(u >> 16);
}
__device__ __forceinline__ float bf16_to_f(unsigned short h) {
  unsigned u = ((unsigned)h) << 16;
  return __builtin_bit_cast(float, u);
}

// async global->LDS b128 DMA (gfx1250, ASYNCcnt-tracked)
__device__ __forceinline__ void async_b128(const unsigned short* g, unsigned short* l) {
  __builtin_amdgcn_global_load_async_to_lds_b128(
      (__attribute__((address_space(1))) i32x4*)(g),
      (__attribute__((address_space(3))) i32x4*)(l), 0, 0);
}
__device__ __forceinline__ void wait_async0() {
  asm volatile("s_wait_asynccnt 0x0" ::: "memory");
}

// ---------------- kernel 1: split W into hi/lo bf16, K-major tiled ----------
extern "C" __global__ void split_weights(const float* __restrict__ w,
                                         unsigned short* __restrict__ wh,
                                         unsigned short* __restrict__ wl) {
  int i = blockIdx.x * blockDim.x + threadIdx.x;
  if (i >= CDIM * KDIM) return;
  float f = w[i];                      // W[n][k] row-major (256,2048)
  unsigned short h = bf16_trunc(f);
  float lf = f - bf16_to_f(h);         // exact residual
  unsigned short l = bf16_trunc(lf);
  int nrow = i >> 11;
  int k    = i & 2047;
  int o = (k >> 5) * (NGEMM * 32) + nrow * 32 + (k & 31);   // [kstep][n][kk]
  wh[o] = h;
  wl[o] = l;
}

// ---------------- kernel 2: scan children -> scatter maps -------------------
extern "C" __global__ void scan_children(const int* __restrict__ ch,
                                         int* __restrict__ leaf_pos,
                                         int* __restrict__ nonleaf_pos) {
  __shared__ int s[1024];
  __shared__ int baseLeaf, baseNon;
  const int tid = threadIdx.x;
  if (tid == 0) { baseLeaf = 0; baseNon = 0; }
  __syncthreads();
  const int n = LEAF_N + NONLEAF_N;
  for (int chunk = 0; chunk < n; chunk += 1024) {
    int idx = chunk + tid;
    int v = (idx < n) ? ((ch[idx] < 0) ? 1 : 0) : 0;
    s[tid] = v;
    __syncthreads();
    #pragma unroll 1
    for (int off = 1; off < 1024; off <<= 1) {
      int t = (tid >= off) ? s[tid - off] : 0;
      __syncthreads();
      s[tid] += t;
      __syncthreads();
    }
    int incl = s[tid];
    int excl = incl - v;
    int bl = baseLeaf, bn = baseNon;
    if (idx < n) {
      if (v) leaf_pos[bl + excl] = idx;
      else   nonleaf_pos[bn + (tid - excl)] = idx;
    }
    __syncthreads();
    if (tid == 0) {
      int total = s[1023];
      int valid = (n - chunk < 1024) ? (n - chunk) : 1024;
      baseLeaf = bl + total;
      baseNon  = bn + (valid - total);
    }
    __syncthreads();
  }
}

// ---------------- kernel 3: prefix + leaf row copies ------------------------
extern "C" __global__ void copy_rows(const float* __restrict__ x,
                                     const int* __restrict__ leaf_pos,
                                     float* __restrict__ out) {
  int gid = blockIdx.x * blockDim.x + threadIdx.x;   // one float4
  int row = gid >> 6;
  int c4  = gid & 63;
  if (row >= PREFIX_N + LEAF_N) return;
  int srcRow, dstRow;
  if (row < PREFIX_N) { srcRow = row; dstRow = row; }
  else {
    int l  = row - PREFIX_N;
    srcRow = PREFIX_N + l;
    dstRow = PREFIX_N + leaf_pos[l];
  }
  const float4* src = (const float4*)(x   + (size_t)srcRow * CDIM) + c4;
  float4*       dst = (float4*)      (out + (size_t)dstRow * CDIM) + c4;
  *dst = *src;
}

// ---------------- kernel 4: bf16x3 WMMA GEMM, async-LDS double-buffered -----
__device__ __forceinline__ v16bf load_a_frag(const unsigned short* s, int row, int sel) {
  // A 16x32 bf16: lanes 0-15: K=0..7 & 16..23 ; lanes 16-31: K=8..15 & 24..31
  uint4 a = *(const uint4*)(s + row * APITCH + sel * 8);
  uint4 b = *(const uint4*)(s + row * APITCH + 16 + sel * 8);
  u32x8 t;
  t[0]=a.x; t[1]=a.y; t[2]=a.z; t[3]=a.w;
  t[4]=b.x; t[5]=b.y; t[6]=b.z; t[7]=b.w;
  return __builtin_bit_cast(v16bf, t);
}
__device__ __forceinline__ v16bf load_b_frag(const unsigned short* s, int row, int sel) {
  // B 32x16 bf16: lanes 0-15: K=0..7 & 16..23 ; lanes 16-31: K=8..15 & 24..31
  uint4 a = *(const uint4*)(s + row * BPITCH + sel * 8);
  uint4 b = *(const uint4*)(s + row * BPITCH + 16 + sel * 8);
  u32x8 t;
  t[0]=a.x; t[1]=a.y; t[2]=a.z; t[3]=a.w;
  t[4]=b.x; t[5]=b.y; t[6]=b.z; t[7]=b.w;
  return __builtin_bit_cast(v16bf, t);
}

extern "C" __global__ void __launch_bounds__(256)
gemm_scatter(const float* __restrict__ xd,
             const unsigned short* __restrict__ wh,
             const unsigned short* __restrict__ wl,
             const int* __restrict__ nonleaf_pos,
             float* __restrict__ out) {
  // carved LDS: A/B double buffers (102400B); epilogue tile reuses same block
  __shared__ __align__(16) char smem[102400];
  unsigned short* sAh = (unsigned short*)smem;            // [2][64*40]   (10240B)
  unsigned short* sAl = sAh + 2 * (MBLK * APITCH);        // [2][64*40]   (10240B)
  unsigned short* sBh = sAl + 2 * (MBLK * APITCH);        // [2][256*40]  (40960B)
  unsigned short* sBl = sBh + 2 * (NGEMM * BPITCH);       // [2][256*40]  (40960B)

  const int tid   = threadIdx.x;
  const int lane  = tid & 31;
  const int wv    = tid >> 5;       // 8 waves
  const int msub  = wv & 3;         // M-subtile of 16
  const int nhalf = wv >> 2;        // N 0-127 / 128-255
  const int lrow  = lane & 15;
  const int sel   = lane >> 4;
  const int mbase = blockIdx.x * MBLK;

  v8f acc[8] = {};

  // A staging assignment: thread -> (row 0..63, 8-float chunk 0..3)
  const int a_row = tid >> 2;
  const int a_ch  = tid & 3;
  int a_m = mbase + a_row;
  if (a_m >= MGEMM) a_m = MGEMM - 1;            // clamp (safe duplicate)
  const float* aptr = xd + (size_t)a_m * KDIM + a_ch * 8;

  auto stageB = [&](int ks, int buf) {          // async DMA global->LDS
    const unsigned short* gh = wh + (size_t)ks * (NGEMM * 32) + tid * 32;
    const unsigned short* gl = wl + (size_t)ks * (NGEMM * 32) + tid * 32;
    unsigned short* dh = sBh + buf * (NGEMM * BPITCH) + tid * BPITCH;
    unsigned short* dl = sBl + buf * (NGEMM * BPITCH) + tid * BPITCH;
    async_b128(gh,      dh);
    async_b128(gh + 8,  dh + 8);
    async_b128(gh + 16, dh + 16);
    async_b128(gh + 24, dh + 24);
    async_b128(gl,      dl);
    async_b128(gl + 8,  dl + 8);
    async_b128(gl + 16, dl + 16);
    async_b128(gl + 24, dl + 24);
  };

  auto stageA = [&](int ks, int buf) {          // fp32 load -> hi/lo bf16 -> LDS
    float4 f0 = *(const float4*)(aptr + ks * 32);
    float4 f1 = *(const float4*)(aptr + ks * 32 + 4);
    float fv[8] = {f0.x, f0.y, f0.z, f0.w, f1.x, f1.y, f1.z, f1.w};
    uint4 uh, ul;
    unsigned hw[4], lw[4];
    #pragma unroll
    for (int i = 0; i < 4; ++i) {
      unsigned short h0 = bf16_trunc(fv[2*i]);
      unsigned short h1 = bf16_trunc(fv[2*i+1]);
      float l0f = fv[2*i]   - bf16_to_f(h0);
      float l1f = fv[2*i+1] - bf16_to_f(h1);
      hw[i] = (unsigned)h0 | ((unsigned)h1 << 16);
      lw[i] = (unsigned)bf16_trunc(l0f) | ((unsigned)bf16_trunc(l1f) << 16);
    }
    uh.x = hw[0]; uh.y = hw[1]; uh.z = hw[2]; uh.w = hw[3];
    ul.x = lw[0]; ul.y = lw[1]; ul.z = lw[2]; ul.w = lw[3];
    *(uint4*)(sAh + buf * (MBLK * APITCH) + a_row * APITCH + a_ch * 8) = uh;
    *(uint4*)(sAl + buf * (MBLK * APITCH) + a_row * APITCH + a_ch * 8) = ul;
  };

  // prologue: fill buffer 0
  stageB(0, 0);
  stageA(0, 0);
  wait_async0();
  __syncthreads();

  for (int ks = 0; ks < KSTEPS; ++ks) {
    const int cur = ks & 1;
    if (ks + 1 < KSTEPS) {
      stageB(ks + 1, cur ^ 1);                  // async DMA into next buffer
      stageA(ks + 1, cur ^ 1);                  // VALU split into next buffer
    }
    if (ks + 2 < KSTEPS)
      __builtin_prefetch(aptr + (ks + 2) * 32, 0, 1);   // global_prefetch_b8

    // compute on current buffer: 8 N-tiles x (Ah*Bh + Al*Bh + Ah*Bl)
    const unsigned short* pAh = sAh + cur * (MBLK * APITCH);
    const unsigned short* pAl = sAl + cur * (MBLK * APITCH);
    const unsigned short* pBh = sBh + cur * (NGEMM * BPITCH);
    const unsigned short* pBl = sBl + cur * (NGEMM * BPITCH);
    v16bf Ah = load_a_frag(pAh, msub * 16 + lrow, sel);
    v16bf Al = load_a_frag(pAl, msub * 16 + lrow, sel);
    #pragma unroll
    for (int nt = 0; nt < 8; ++nt) {
      int brow = nhalf * 128 + nt * 16 + lrow;
      v16bf Bh = load_b_frag(pBh, brow, sel);
      v16bf Bl = load_b_frag(pBl, brow, sel);
      acc[nt] = __builtin_amdgcn_wmma_f32_16x16x32_bf16(false, Ah, false, Bh,
                                                        (short)0, acc[nt], false, false);
      acc[nt] = __builtin_amdgcn_wmma_f32_16x16x32_bf16(false, Al, false, Bh,
                                                        (short)0, acc[nt], false, false);
      acc[nt] = __builtin_amdgcn_wmma_f32_16x16x32_bf16(false, Ah, false, Bl,
                                                        (short)0, acc[nt], false, false);
    }

    wait_async0();      // own wave's async DMAs into next buffer are done
    __syncthreads();    // dscnt-drained barrier: everyone done reading cur / writing nxt
  }

  // ---- epilogue: accumulators -> LDS tile -> coalesced 1KB row scatter ----
  float* sC = (float*)smem;                     // [64][CPITCH] floats (66560B <= 102400B)
  #pragma unroll
  for (int nt = 0; nt < 8; ++nt) {
    #pragma unroll
    for (int j = 0; j < 8; ++j) {
      int r = msub * 16 + sel * 8 + j;          // C/D layout: VGPR j -> M = j + 8*sel
      int c = nhalf * 128 + nt * 16 + lrow;
      sC[r * CPITCH + c] = acc[nt][j];
    }
  }
  __syncthreads();
  #pragma unroll 1
  for (int i = 0; i < 16; ++i) {
    int flat = tid + i * 256;                   // 0..4095 -> (row, float4)
    int r  = flat >> 6;
    int c4 = flat & 63;
    int m  = mbase + r;
    if (m < MGEMM) {
      int dr = PREFIX_N + nonleaf_pos[m];
      float4 v = *(const float4*)&sC[r * CPITCH + c4 * 4];
      *((float4*)(out + (size_t)dr * CDIM) + c4) = v;
    }
  }
}

// ---------------- launcher --------------------------------------------------
extern "C" void kernel_launch(void* const* d_in, const int* in_sizes, int n_in,
                              void* d_out, int out_size, void* d_ws, size_t ws_size,
                              hipStream_t stream) {
  const float* x        = (const float*)d_in[0];
  const float* w        = (const float*)d_in[1];
  const int*   children = (const int*)d_in[2];
  float*       out      = (float*)d_out;

  char* ws = (char*)d_ws;
  unsigned short* wh          = (unsigned short*)(ws);                       // 1 MB
  unsigned short* wl          = (unsigned short*)(ws + (1u << 20));          // 1 MB
  int*            leaf_pos    = (int*)(ws + (2u << 20));                     // 200 KB
  int*            nonleaf_pos = (int*)(ws + (2u << 20) + 204800);            // 200 KB

  split_weights<<<(CDIM * KDIM) / 256, 256, 0, stream>>>(w, wh, wl);
  scan_children<<<1, 1024, 0, stream>>>(children, leaf_pos, nonleaf_pos);
  copy_rows<<<((PREFIX_N + LEAF_N) * (CDIM / 4) + 255) / 256, 256, 0, stream>>>(
      x, leaf_pos, out);
  const float* xd = x + (size_t)(PREFIX_N + LEAF_N) * CDIM;
  gemm_scatter<<<(MGEMM + MBLK - 1) / MBLK, 256, 0, stream>>>(
      xd, wh, wl, nonleaf_pos, out);
}
